// KGEModel_55430847922741
// MI455X (gfx1250) — matrix operand order repaired and checked by reading, not compiled, not run on hardware
//
#include <hip/hip_runtime.h>

#define BATCH 8192
#define DIM   256
#define SEQ   31
#define SEQP  32
#define NBR   10
#define ANCH  20
#define HEADS 8
#define HD    32
#define NL    2
#define FF    1024

typedef _Float16 f16;
typedef __attribute__((ext_vector_type(16))) _Float16 v16h;
typedef __attribute__((ext_vector_type(8)))  _Float16 h8;
typedef __attribute__((ext_vector_type(8)))  float    v8f;

// ---- WMMA fragment helpers (gfx1250, wave32, 16x16x32 f16) ----
// A (16x32, 16-bit): lane m=lane&15; elems 0..15 = K (hi?16:0)+0..15 contiguous.
__device__ __forceinline__ v16h frag_a(const f16* base, int stride, int row0, int k0, int lane) {
  int r = lane & 15;
  int ho = (lane >> 4) << 4;                 // 0 or 16 halfs
  const f16* p = base + (size_t)(row0 + r) * stride + k0 + ho;
  h8 lo = *(const h8*)p;
  h8 hi = *(const h8*)(p + 8);
  return __builtin_shufflevector(lo, hi, 0,1,2,3,4,5,6,7,8,9,10,11,12,13,14,15);
}
// B (32x16, 16-bit): lane n=lane&15 holds column n, K contiguous; lanes>=16 take K+16.
// p must already point at (column row) + k0 + ((lane>>4)<<4).
__device__ __forceinline__ v16h frag_b(const f16* p) {
  h8 lo = *(const h8*)p;
  h8 hi = *(const h8*)(p + 8);
  return __builtin_shufflevector(lo, hi, 0,1,2,3,4,5,6,7,8,9,10,11,12,13,14,15);
}
__device__ __forceinline__ v8f wmma16(v16h a, v16h b, v8f c) {
  return __builtin_amdgcn_wmma_f32_16x16x32_f16(false, a, false, b, (short)0, c, false, false);
}
__device__ __forceinline__ float gelu_tanh(float x) {
  float x3 = x * x * x;
  return 0.5f * x * (1.0f + tanhf(0.7978845608028654f * (x + 0.044715f * x3)));
}

// ---- weight prep: f32 -> f16, transposed to [l][N][K] row-major ----
__global__ void prep_weights(const float* wq, const float* wk, const float* wv, const float* wo,
                             const float* w1, const float* w2,
                             f16* wqT, f16* wkT, f16* wvT, f16* woT, f16* w1T, f16* w2T) {
  int idx = blockIdx.x * blockDim.x + threadIdx.x;          // 0 .. 524287
  if (idx < NL * DIM * DIM) {
    int l = idx >> 16, rem = idx & 65535;
    int n = rem >> 8, k = rem & 255;
    int src = (l * DIM + k) * DIM + n;
    wqT[idx] = (f16)wq[src];
    wkT[idx] = (f16)wk[src];
    wvT[idx] = (f16)wv[src];
    woT[idx] = (f16)wo[src];
  }
  if (idx < NL * FF * DIM) {
    int l = idx >> 18, rem = idx & 262143;
    { int n = rem >> 8;  int k = rem & 255;  w1T[idx] = (f16)w1[(l * DIM + k) * FF  + n]; }
    { int n = rem >> 10; int k = rem & 1023; w2T[idx] = (f16)w2[(l * FF  + k) * DIM + n]; }
  }
}

// ---- embedding gather + path modulation -> x[B][32][256] f32 ----
__global__ void embed_kernel(const int* entities, const int* hashes, const int* nodes,
                             const int* path_ids, const float* anchor_emb, const float* node_emb,
                             const float* path_emb, const float* type_emb, float* x) {
  int bb = blockIdx.x;
  int c  = threadIdx.x;
  int e  = entities[bb];
  float* xb = x + (size_t)bb * SEQP * DIM;
  for (int r = 0; r < 1 + NBR; ++r) {
    int nid = nodes[e * (1 + NBR) + r];
    xb[r * DIM + c] = node_emb[(size_t)nid * DIM + c] + type_emb[(r == 0 ? 0 : 1) * DIM + c];
  }
  float t2 = type_emb[2 * DIM + c];
  for (int a = 0; a < ANCH; ++a) {
    int hsh = hashes[e * ANCH + a];
    float v = anchor_emb[(size_t)hsh * DIM + c];
    int p0 = path_ids[(e * ANCH + a) * 2 + 0];
    int p1 = path_ids[(e * ANCH + a) * 2 + 1];
    const float* pe0 = path_emb + (size_t)p0 * (2 * DIM);
    v = v * (pe0[c] + 1.0f) + pe0[DIM + c];
    const float* pe1 = path_emb + (size_t)p1 * (2 * DIM);
    v = v * (pe1[c] + 1.0f) + pe1[DIM + c];
    xb[(1 + NBR + a) * DIM + c] = v + t2;
  }
  xb[31 * DIM + c] = 0.0f;                                  // pad row
}

// ---- attention block: 2 entities / block, 256 threads (8 waves) ----
__global__ __launch_bounds__(256) void attn_layer(const float* ln1_w, const float* ln1_b,
                                                  const f16* wqT, const f16* wkT,
                                                  const f16* wvT, const f16* woT,
                                                  float* x, int l) {
  __shared__ f16 sm_hatt[20480];     // h: [64][272] (34.8KB)  ALIASED with att: [2][8][32][40]
  __shared__ f16 sm_q[64 * 272];     // Q rows, later ctx rows
  __shared__ f16 sm_k[64 * 272];     // K rows (serve as B fragments of Q@K^T)
  __shared__ f16 sm_vt[256 * 72];    // V transposed: [d][seq0..63]

  const int t    = threadIdx.x;
  const int lane = t & 31;
  const int wvid = t >> 5;
  const int nn   = lane & 15;
  const int mb   = (lane >> 4) << 3;
  const int hi16 = (lane >> 4) << 4;
  const size_t xbase = (size_t)blockIdx.x * 64 * DIM;

  // LN1: 4 lanes per row, shfl reduce
  {
    int row = t >> 2, qd = t & 3;
    const float* xr = x + xbase + row * DIM + qd * 64;
    float s = 0.f, sq = 0.f;
    for (int i = 0; i < 64; i += 4) {
      float4 v = *(const float4*)(xr + i);
      s  += v.x + v.y + v.z + v.w;
      sq += v.x * v.x + v.y * v.y + v.z * v.z + v.w * v.w;
    }
    s += __shfl_xor(s, 1, 32);  sq += __shfl_xor(sq, 1, 32);
    s += __shfl_xor(s, 2, 32);  sq += __shfl_xor(sq, 2, 32);
    float m  = s * (1.0f / DIM);
    float rs = rsqrtf(sq * (1.0f / DIM) - m * m + 1e-5f);
    const float* lw = ln1_w + l * DIM + qd * 64;
    const float* lb = ln1_b + l * DIM + qd * 64;
    f16* hr = sm_hatt + row * 272 + qd * 64;
    for (int i = 0; i < 64; ++i) hr[i] = (f16)((xr[i] - m) * rs * lw[i] + lb[i]);
  }
  __syncthreads();

  // QKV GEMM: [64x256] @ [256x256] x3 -> 192 tiles
  for (int tid = wvid; tid < 192; tid += 8) {
    int which = tid >> 6, tt = tid & 63;
    int mt = tt >> 4, nt = tt & 15;
    const f16* WT = (which == 0 ? wqT : which == 1 ? wkT : wvT) + l * DIM * DIM;
    v8f acc = {0.f, 0.f, 0.f, 0.f, 0.f, 0.f, 0.f, 0.f};
    for (int k0 = 0; k0 < DIM; k0 += 32) {
      v16h a = frag_a(sm_hatt, 272, mt * 16, k0, lane);
      v16h b = frag_b(WT + (nt * 16 + nn) * DIM + k0 + hi16);
      acc = wmma16(a, b, acc);
    }
    if (which == 2) {
      for (int j = 0; j < 8; ++j)
        sm_vt[(nt * 16 + nn) * 72 + mt * 16 + mb + j] = (f16)acc[j];
    } else {
      f16* dst = (which == 0) ? sm_q : sm_k;
      for (int j = 0; j < 8; ++j)
        dst[(mt * 16 + mb + j) * 272 + nt * 16 + nn] = (f16)acc[j];
    }
  }
  __syncthreads();

  // scores: per (entity, head) 32x32 = Q_h @ K_h^T, scaled, pad col masked
  for (int tid = wvid; tid < 64; tid += 8) {
    int i = tid >> 5, rem = tid & 31;
    int hh = rem >> 2, mt = (rem >> 1) & 1, nt = rem & 1;
    v8f acc = {0.f, 0.f, 0.f, 0.f, 0.f, 0.f, 0.f, 0.f};
    v16h a = frag_a(sm_q, 272, i * 32 + mt * 16, hh * 32, lane);
    v16h b = frag_b(sm_k + (i * 32 + nt * 16 + nn) * 272 + hh * 32 + hi16);
    acc = wmma16(a, b, acc);
    int kcol = nt * 16 + nn;
    for (int j = 0; j < 8; ++j) {
      float sv = (kcol == 31) ? -60000.0f : (float)acc[j] * 0.17677669529663687f;
      sm_hatt[((i * 8 + hh) * 32 + mt * 16 + mb + j) * 40 + kcol] = (f16)sv;
    }
  }
  __syncthreads();

  // softmax over 512 rows of 32
  for (int rr = t; rr < 512; rr += 256) {
    f16* rp = sm_hatt + rr * 40;
    float v[32], mx = -1e30f;
    for (int c = 0; c < 32; ++c) { v[c] = (float)rp[c]; mx = fmaxf(mx, v[c]); }
    float ssum = 0.f;
    for (int c = 0; c < 32; ++c) { float e = __expf(v[c] - mx); v[c] = e; ssum += e; }
    float inv = 1.0f / ssum;
    for (int c = 0; c < 32; ++c) rp[c] = (f16)(v[c] * inv);
  }
  __syncthreads();

  // ctx = att @ V  (B fragments from V^T, K = seq)
  for (int tid = wvid; tid < 64; tid += 8) {
    int i = tid >> 5, rem = tid & 31;
    int hh = rem >> 2, mt = (rem >> 1) & 1, nt = rem & 1;
    v8f acc = {0.f, 0.f, 0.f, 0.f, 0.f, 0.f, 0.f, 0.f};
    v16h a = frag_a(sm_hatt, 40, (i * 8 + hh) * 32 + mt * 16, 0, lane);
    v16h b = frag_b(sm_vt + (hh * 32 + nt * 16 + nn) * 72 + i * 32 + hi16);
    acc = wmma16(a, b, acc);
    for (int j = 0; j < 8; ++j)
      sm_q[(i * 32 + mt * 16 + mb + j) * 272 + hh * 32 + nt * 16 + nn] = (f16)acc[j];
  }
  __syncthreads();

  // out projection + f32 residual
  const f16* WT = woT + l * DIM * DIM;
  for (int tid = wvid; tid < 64; tid += 8) {
    int mt = tid >> 4, nt = tid & 15;
    v8f acc = {0.f, 0.f, 0.f, 0.f, 0.f, 0.f, 0.f, 0.f};
    for (int k0 = 0; k0 < DIM; k0 += 32) {
      v16h a = frag_a(sm_q, 272, mt * 16, k0, lane);
      v16h b = frag_b(WT + (nt * 16 + nn) * DIM + k0 + hi16);
      acc = wmma16(a, b, acc);
    }
    for (int j = 0; j < 8; ++j) {
      size_t o = xbase + (mt * 16 + mb + j) * DIM + nt * 16 + nn;
      x[o] += (float)acc[j];
    }
  }
}

// ---- MLP block: LN2 -> gelu(h@W1+b1)@W2 + b2, hidden split into 2x512 cols ----
__global__ __launch_bounds__(256) void mlp_layer(const float* ln2_w, const float* ln2_b,
                                                 const f16* w1T, const f16* w2T,
                                                 const float* b1, const float* b2,
                                                 float* x, int l) {
  __shared__ f16 sm_h[64 * 272];     // LN output
  __shared__ f16 sm_h2[64 * 528];    // gelu hidden, one 512-col half at a time

  const int t    = threadIdx.x;
  const int lane = t & 31;
  const int wvid = t >> 5;
  const int nn   = lane & 15;
  const int mb   = (lane >> 4) << 3;
  const int hi16 = (lane >> 4) << 4;
  const size_t xbase = (size_t)blockIdx.x * 64 * DIM;

  {
    int row = t >> 2, qd = t & 3;
    const float* xr = x + xbase + row * DIM + qd * 64;
    float s = 0.f, sq = 0.f;
    for (int i = 0; i < 64; i += 4) {
      float4 v = *(const float4*)(xr + i);
      s  += v.x + v.y + v.z + v.w;
      sq += v.x * v.x + v.y * v.y + v.z * v.z + v.w * v.w;
    }
    s += __shfl_xor(s, 1, 32);  sq += __shfl_xor(sq, 1, 32);
    s += __shfl_xor(s, 2, 32);  sq += __shfl_xor(sq, 2, 32);
    float m  = s * (1.0f / DIM);
    float rs = rsqrtf(sq * (1.0f / DIM) - m * m + 1e-5f);
    const float* lw = ln2_w + l * DIM + qd * 64;
    const float* lb = ln2_b + l * DIM + qd * 64;
    f16* hr = sm_h + row * 272 + qd * 64;
    for (int i = 0; i < 64; ++i) hr[i] = (f16)((xr[i] - m) * rs * lw[i] + lb[i]);
  }
  __syncthreads();

  v8f zero = {0.f, 0.f, 0.f, 0.f, 0.f, 0.f, 0.f, 0.f};
  v8f oacc[8];
  for (int p = 0; p < 8; ++p) oacc[p] = zero;

  for (int half = 0; half < 2; ++half) {
    // GEMM1 [64x256]@[256x512] + bias + gelu -> sm_h2
    for (int tid = wvid; tid < 128; tid += 8) {
      int mt = tid >> 5, nt = tid & 31;
      v8f acc = zero;
      const f16* W = w1T + (size_t)(l * FF + half * 512 + nt * 16 + nn) * DIM;
      for (int k0 = 0; k0 < DIM; k0 += 32) {
        v16h a = frag_a(sm_h, 272, mt * 16, k0, lane);
        v16h b = frag_b(W + k0 + hi16);
        acc = wmma16(a, b, acc);
      }
      int ncol = nt * 16 + nn;
      float bb = b1[l * FF + half * 512 + ncol];
      for (int j = 0; j < 8; ++j)
        sm_h2[(mt * 16 + mb + j) * 528 + ncol] = (f16)gelu_tanh((float)acc[j] + bb);
    }
    __syncthreads();
    // GEMM2 [64x512]@[512x256] accumulate into register fragments
    for (int p = 0; p < 8; ++p) {
      int tl = wvid + 8 * p;
      int mt = tl >> 4, nt = tl & 15;
      const f16* W = w2T + (size_t)(l * DIM + nt * 16 + nn) * FF + half * 512;
      for (int k0 = 0; k0 < 512; k0 += 32) {
        v16h a = frag_a(sm_h2, 528, mt * 16, k0, lane);
        v16h b = frag_b(W + k0 + hi16);
        oacc[p] = wmma16(a, b, oacc[p]);
      }
    }
    __syncthreads();
  }

  for (int p = 0; p < 8; ++p) {
    int tl = wvid + 8 * p;
    int mt = tl >> 4, nt = tl & 15;
    for (int j = 0; j < 8; ++j) {
      int ncol = nt * 16 + nn;
      size_t o = xbase + (mt * 16 + mb + j) * DIM + ncol;
      x[o] += (float)oacc[p][j] + b2[l * DIM + ncol];
    }
  }
}

// ---- mean pool over 31 real rows ----
__global__ void pool_kernel(const float* x, float* out) {
  int bb = blockIdx.x, c = threadIdx.x;
  const float* xb = x + (size_t)bb * SEQP * DIM;
  float s = 0.f;
  for (int r = 0; r < SEQ; ++r) s += xb[r * DIM + c];
  out[(size_t)bb * DIM + c] = s * (1.0f / SEQ);
}

extern "C" void kernel_launch(void* const* d_in, const int* in_sizes, int n_in,
                              void* d_out, int out_size, void* d_ws, size_t ws_size,
                              hipStream_t stream) {
  const int*   entities   = (const int*)d_in[0];
  const int*   hashes     = (const int*)d_in[1];
  const int*   nodes      = (const int*)d_in[2];
  const int*   path_ids   = (const int*)d_in[3];
  /* d_in[4] type_ids: fixed [0, 1*10, 2*20] — baked into embed_kernel */
  const float* anchor_emb = (const float*)d_in[5];
  const float* node_emb   = (const float*)d_in[6];
  const float* path_emb   = (const float*)d_in[7];
  const float* type_emb   = (const float*)d_in[8];
  const float* ln1_w = (const float*)d_in[9];
  const float* ln1_b = (const float*)d_in[10];
  const float* wq    = (const float*)d_in[11];
  const float* wk    = (const float*)d_in[12];
  const float* wv    = (const float*)d_in[13];
  const float* wo    = (const float*)d_in[14];
  const float* ln2_w = (const float*)d_in[15];
  const float* ln2_b = (const float*)d_in[16];
  const float* w1    = (const float*)d_in[17];
  const float* b1    = (const float*)d_in[18];
  const float* w2    = (const float*)d_in[19];
  const float* b2    = (const float*)d_in[20];

  char* ws = (char*)d_ws;
  float* x = (float*)ws;
  size_t off = (size_t)BATCH * SEQP * DIM * sizeof(float);
  f16* wqT = (f16*)(ws + off); off += (size_t)NL * DIM * DIM * 2;
  f16* wkT = (f16*)(ws + off); off += (size_t)NL * DIM * DIM * 2;
  f16* wvT = (f16*)(ws + off); off += (size_t)NL * DIM * DIM * 2;
  f16* woT = (f16*)(ws + off); off += (size_t)NL * DIM * DIM * 2;
  f16* w1T = (f16*)(ws + off); off += (size_t)NL * DIM * FF * 2;
  f16* w2T = (f16*)(ws + off); off += (size_t)NL * DIM * FF * 2;

  prep_weights<<<2048, 256, 0, stream>>>(wq, wk, wv, wo, w1, w2, wqT, wkT, wvT, woT, w1T, w2T);
  embed_kernel<<<BATCH, 256, 0, stream>>>(entities, hashes, nodes, path_ids,
                                          anchor_emb, node_emb, path_emb, type_emb, x);
  for (int l = 0; l < NL; ++l) {
    attn_layer<<<BATCH / 2, 256, 0, stream>>>(ln1_w, ln1_b, wqT, wkT, wvT, woT, x, l);
    mlp_layer<<<BATCH / 2, 256, 0, stream>>>(ln2_w, ln2_b, w1T, w2T, b1, b2, x, l);
  }
  pool_kernel<<<BATCH, 256, 0, stream>>>(x, (float*)d_out);
}
